// Sindy_53815940219534
// MI455X (gfx1250) — matrix operand order repaired and checked by reading, not compiled, not run on hardware
//
#include <hip/hip_runtime.h>

#define D               16
#define LIB             969
#define LIBP            992          // 31 * 32 (K padded)
#define NCHUNK          31
#define WAVES           4
#define ROWS_PER_WAVE   32
#define ROWS_PER_BLOCK  (WAVES * ROWS_PER_WAVE)   // 128
#define ABUF_STRIDE     144          // bytes per staged A row (64B data + pad)
#define ABUF_BYTES      (ROWS_PER_WAVE * ABUF_STRIDE)       // 4608
#define BSTAGE_BYTES    (NCHUNK * 1024)                      // 31744
#define LDS_TOTAL       (BSTAGE_BYTES + WAVES * ABUF_BYTES)  // 50176

typedef _Float16     v16h __attribute__((ext_vector_type(16)));
typedef float        v8f  __attribute__((ext_vector_type(8)));
typedef unsigned int v8u  __attribute__((ext_vector_type(8)));

// ---- compile-time feature table (permuted order + canonical index for B staging) ----
struct PermEntry { int ord, a, b, c, canon; };
struct Tables { PermEntry e[LIBP]; };

constexpr Tables build_tables() {
  Tables t{};
  int qidx[D][D] = {};
  int cidx[D][D][D] = {};
  int f = 1 + D;
  for (int i = 0; i < D; i++)
    for (int j = i; j < D; j++) qidx[i][j] = f++;
  for (int i = 0; i < D; i++)
    for (int j = i; j < D; j++)
      for (int k = j; k < D; k++) cidx[i][j][k] = f++;
  int p = 0;
  t.e[p++] = PermEntry{0, 0, 0, 0, 0};                       // constant 1
  for (int i = 0; i < D; i++) { t.e[p] = PermEntry{1, i, 0, 0, 1 + i}; p++; }
  for (int i = 0; i < D; i++)                                // quad followed by its cubics
    for (int j = i; j < D; j++) {
      t.e[p] = PermEntry{2, i, j, 0, qidx[i][j]}; p++;
      for (int k = j; k < D; k++) { t.e[p] = PermEntry{3, i, j, k, cidx[i][j][k]}; p++; }
    }
  for (; p < LIBP; p++) t.e[p] = PermEntry{4, 0, 0, 0, LIB}; // zero pad
  return t;
}

constexpr Tables cTBL = build_tables();          // compile-time only (template reads)

struct CanonArr { unsigned short v[LIBP]; };
constexpr CanonArr build_canon() {
  CanonArr c{};
  Tables t = build_tables();
  for (int i = 0; i < LIBP; i++) c.v[i] = (unsigned short)t.e[i].canon;
  return c;
}
__device__ const CanonArr g_canon = build_canon();  // runtime-indexed (B staging only)

static __device__ __forceinline__ v16h assemble_frag(uint4 lo, uint4 hi) {
  v8u u;
  u[0] = lo.x; u[1] = lo.y; u[2] = lo.z; u[3] = lo.w;
  u[4] = hi.x; u[5] = hi.y; u[6] = hi.z; u[7] = hi.w;
  return __builtin_bit_cast(v16h, u);
}

// one hardware instruction: v_cvt_pk_rtz_f16_f32
static __device__ __forceinline__ unsigned pack_rtz(float lo, float hi) {
  auto p = __builtin_amdgcn_cvt_pkrtz(lo, hi);   // __fp16 ext_vector(2)
  return __builtin_bit_cast(unsigned, p);
}

struct KState {
  float r[D];
  float pq;
  unsigned pk0, pk1, pk2, pk3;
  v8f acc0, acc1;
  unsigned char*       myArow;   // this lane's A staging row (store target)
  const unsigned char* aload0;   // A frag tile0 base: row (lane&15) + halfsel
  const unsigned char* aload1;   // A frag tile1 base: row (lane&15)+16 + halfsel
  const unsigned char* bbase;    // B stage base + lane*32
};

template <int P>
static __device__ __forceinline__ float featval(KState& s) {
  constexpr PermEntry e = cTBL.e[P];   // forced compile-time
  if constexpr (e.ord == 0)      return 1.0f;
  else if constexpr (e.ord == 1) return s.r[e.a];
  else if constexpr (e.ord == 2) { s.pq = s.r[e.a] * s.r[e.b]; return s.pq; }
  else if constexpr (e.ord == 3) return s.pq * s.r[e.c];
  else                           return 0.0f;
}

template <int PP>   // pair index: features 2*PP and 2*PP+1
static __device__ __forceinline__ void emit_pair(KState& s) {
  float v0 = featval<2 * PP>(s);
  float v1 = featval<2 * PP + 1>(s);
  unsigned u = pack_rtz(v0, v1);

  constexpr int t    = (2 * PP + 1) & 31;   // odd position within chunk
  constexpr int slot = (t >> 1) & 3;
  if constexpr (slot == 0)      s.pk0 = u;
  else if constexpr (slot == 1) s.pk1 = u;
  else if constexpr (slot == 2) s.pk2 = u;
  else                          s.pk3 = u;

  if constexpr ((t & 7) == 7) {
    *(uint4*)(s.myArow + (t >> 3) * 16) = make_uint4(s.pk0, s.pk1, s.pk2, s.pk3); // ds_store_b128
  }

  if constexpr (t == 31) {
    constexpr int c = (2 * PP) >> 5;        // literal -> folds into ds_load offsets
    uint4 a0l = *(const uint4*)(s.aload0);
    uint4 a0h = *(const uint4*)(s.aload0 + 32);
    uint4 a1l = *(const uint4*)(s.aload1);
    uint4 a1h = *(const uint4*)(s.aload1 + 32);
    uint4 b0  = *(const uint4*)(s.bbase + c * 1024);
    uint4 b1  = *(const uint4*)(s.bbase + c * 1024 + 16);
    v16h A0 = assemble_frag(a0l, a0h);
    v16h A1 = assemble_frag(a1l, a1h);
    v16h Bv = assemble_frag(b0, b1);
    s.acc0 = __builtin_amdgcn_wmma_f32_16x16x32_f16(false, A0, false, Bv, (short)0, s.acc0, false, false);
    s.acc1 = __builtin_amdgcn_wmma_f32_16x16x32_f16(false, A1, false, Bv, (short)0, s.acc1, false, false);
  }
}

template <int Lo, int N>   // over pair indices
static __device__ __forceinline__ void emit_range(KState& s) {
  if constexpr (N == 1) {
    emit_pair<Lo>(s);
  } else {
    constexpr int H = N / 2;
    emit_range<Lo, H>(s);
    emit_range<Lo + H, N - H>(s);
  }
}

__global__ __launch_bounds__(ROWS_PER_BLOCK)
void sindy_kernel(const float* __restrict__ z,
                  const float* __restrict__ coeff,
                  const float* __restrict__ fixedv,
                  const unsigned char* __restrict__ fixedm,
                  float* __restrict__ out, int Btot) {
  __shared__ __align__(16) unsigned char lds[LDS_TOTAL];

  const int tid  = threadIdx.x;
  const int lane = tid & 31;
  const int wave = tid >> 5;
  const int col  = lane & 15;
  const bool hiH = (lane & 16) != 0;

  // ---- one-time: stage coefficients -> f16, WMMA B-fragment swizzled, K-permuted ----
  // B layout: VGPR t, lane L: pair K = 32c + (L<16?0:8) + 2t + (t>=4?8:0) (+1), N = L&15.
  for (int e = tid; e < NCHUNK * 256; e += ROWS_PER_BLOCK) {
    int c = e >> 8;
    int L = (e >> 3) & 31;
    int t = e & 7;
    int K = c * 32 + ((L & 16) ? 8 : 0) + 2 * t + ((t & 4) ? 8 : 0);
    float lo = 0.f, hi = 0.f;
    int can0 = g_canon.v[K];
    if (can0 < LIB) {
      int idx = can0 * 16 + (L & 15);
      lo = fixedm[idx] ? fixedv[idx] : coeff[idx];
    }
    int can1 = g_canon.v[K + 1];
    if (can1 < LIB) {
      int idx = can1 * 16 + (L & 15);
      hi = fixedm[idx] ? fixedv[idx] : coeff[idx];
    }
    *(unsigned int*)(lds + c * 1024 + L * 32 + t * 4) = pack_rtz(lo, hi);
  }
  __syncthreads();

  // ---- per-lane z row (32 distinct rows per wave => no redundant feature compute) ----
  const int rowBase = blockIdx.x * ROWS_PER_BLOCK + wave * ROWS_PER_WAVE;
  const int myRow   = rowBase + lane;

  KState s;
  if (myRow < Btot) {
    const float4* zp = (const float4*)(z + (size_t)myRow * D);
    float4 a0 = zp[0], a1 = zp[1], a2 = zp[2], a3 = zp[3];
    s.r[0]  = a0.x; s.r[1]  = a0.y; s.r[2]  = a0.z; s.r[3]  = a0.w;
    s.r[4]  = a1.x; s.r[5]  = a1.y; s.r[6]  = a1.z; s.r[7]  = a1.w;
    s.r[8]  = a2.x; s.r[9]  = a2.y; s.r[10] = a2.z; s.r[11] = a2.w;
    s.r[12] = a3.x; s.r[13] = a3.y; s.r[14] = a3.z; s.r[15] = a3.w;
  } else {
#pragma unroll
    for (int i = 0; i < D; i++) s.r[i] = 0.f;
  }

  unsigned char* abuf = lds + BSTAGE_BYTES + wave * ABUF_BYTES;
  const int hsel = hiH ? 16 : 0;
  s.myArow = abuf + lane * ABUF_STRIDE;
  s.aload0 = abuf + (lane & 15) * ABUF_STRIDE + hsel;
  s.aload1 = abuf + ((lane & 15) + 16) * ABUF_STRIDE + hsel;
  s.bbase  = lds + lane * 32;
  s.pq = 0.f;
  s.pk0 = s.pk1 = s.pk2 = s.pk3 = 0u;
  s.acc0 = v8f{};
  s.acc1 = v8f{};

  // ---- straight-line stream: pk_mul pairs -> cvt_pk_rtz -> ds_store_b128 -> WMMA per 32 K ----
  emit_range<0, LIBP / 2>(s);

  // ---- epilogue: C/D layout (VGPR v: M = v + (lane<16?0:8), N = lane&15) ----
#pragma unroll
  for (int v = 0; v < 8; ++v) {
    int m  = (hiH ? 8 : 0) + v;
    int r0 = rowBase + m;
    if (r0 < Btot) out[(size_t)r0 * D + col] = s.acc0[v];
    int r1 = rowBase + 16 + m;
    if (r1 < Btot) out[(size_t)r1 * D + col] = s.acc1[v];
  }
}

extern "C" void kernel_launch(void* const* d_in, const int* in_sizes, int n_in,
                              void* d_out, int out_size, void* d_ws, size_t ws_size,
                              hipStream_t stream) {
  const float*         zz     = (const float*)d_in[0];
  const float*         coeff  = (const float*)d_in[1];
  const float*         fixedv = (const float*)d_in[2];
  const unsigned char* fixedm = (const unsigned char*)d_in[3];
  float*               out    = (float*)d_out;
  int Btot   = in_sizes[0] / D;
  int blocks = (Btot + ROWS_PER_BLOCK - 1) / ROWS_PER_BLOCK;
  sindy_kernel<<<blocks, ROWS_PER_BLOCK, 0, stream>>>(zz, coeff, fixedv, fixedm, out, Btot);
}
